// CircularRelativePositionAttention_2422361555015
// MI455X (gfx1250) — matrix operand order repaired and checked by reading, not compiled, not run on hardware
//
#include <hip/hip_runtime.h>

typedef __attribute__((ext_vector_type(16))) _Float16 v16h;
typedef __attribute__((ext_vector_type(4)))  _Float16 v4h;
typedef __attribute__((ext_vector_type(8)))  float    v8f;
typedef __attribute__((ext_vector_type(4)))  float    v4f;

#define SEQ   1000   // sequence length
#define SEQP  1008   // keys padded to 16
#define DM    512    // d_model
#define DH    64     // head dim
#define QTN   63     // query tiles of 16
#define KTI   63     // key tiles of 16
#define SPAD  1024   // padded key dim (multiple of 32)
#define NDIST 512    // padded distance dim (>= 501, multiple of 32)
#define NW    4      // waves per block

// workspace layout (in _Float16 elements)
#define WS_KH   ((size_t)0)                         // [8][1008][512]  f16 K, padded
#define WS_VT   ((size_t)8 * SEQP * DM)             // [64][64][1024]  f16 V^T per (b,h)
#define WS_RKH  (WS_VT + (size_t)64 * DH * SPAD)    // [512][64]       f16 rel_k
#define WS_RVT  (WS_RKH + (size_t)NDIST * DH)       // [64][512]       f16 rel_v^T

__device__ __forceinline__ int circ_dist(int i, int j) {
  int t = (i - j) % SEQ;
  if (t < 0) t += SEQ;
  return min(t, SEQ - t);
}

__device__ __forceinline__ v8f wmma16(v16h a, v16h b, v8f c) {
  return __builtin_amdgcn_wmma_f32_16x16x32_f16(false, a, false, b,
                                                (short)0, c, false, false);
}

// ---------------- prep kernels: build f16 / transposed operand copies -------
__global__ void prep_k_kernel(const float* __restrict__ K, _Float16* __restrict__ ws) {
  int idx = blockIdx.x * blockDim.x + threadIdx.x;       // 8*1008*512
  int b = idx / (SEQP * DM);
  int r = (idx / DM) % SEQP;
  int d = idx % DM;
  float v = (r < SEQ) ? K[((size_t)b * SEQ + r) * DM + d] : 0.0f;
  ws[WS_KH + idx] = (_Float16)v;
}

__global__ void prep_vt_kernel(const float* __restrict__ V, _Float16* __restrict__ ws) {
  int idx = blockIdx.x * blockDim.x + threadIdx.x;       // 64*64*1024
  int bh  = idx >> 16;
  int dim = (idx >> 10) & 63;
  int key = idx & (SPAD - 1);
  int b = bh >> 3, h = bh & 7;
  float v = (key < SEQ) ? V[((size_t)b * SEQ + key) * DM + h * DH + dim] : 0.0f;
  ws[WS_VT + idx] = (_Float16)v;
}

__global__ void prep_rel_kernel(const float* __restrict__ RK,
                                const float* __restrict__ RV,
                                _Float16* __restrict__ ws) {
  int idx = blockIdx.x * blockDim.x + threadIdx.x;       // 2*512*64
  if (idx < NDIST * DH) {
    ws[WS_RKH + idx] = (_Float16)RK[idx];                // rows 0..511 direct copy
  } else {
    int i2 = idx - NDIST * DH;
    int dist = i2 >> 6, d = i2 & 63;
    ws[WS_RVT + (size_t)d * NDIST + dist] = (_Float16)RV[i2];
  }
}

// ---------------- fused attention kernel ------------------------------------
__global__ __launch_bounds__(128, 1)
void circ_rel_attn_kernel(const float* __restrict__ Q,
                          const _Float16* __restrict__ ws,
                          float* __restrict__ Out)
{
  __shared__ _Float16 sQ[16 * DH];       //  2 KB
  __shared__ _Float16 sP[16 * NDIST];    // 16 KB  P[q][dist] = q . rel_k[dist]
  __shared__ float    sS[16 * SPAD];     // 64 KB  scores -> exp(scores)
  __shared__ _Float16 sA[16 * SPAD];     // 32 KB  attn (unnormalized) f16
  __shared__ float    sD[16 * NDIST];    // 32 KB  per-distance attn mass
  __shared__ float    sRed[16 * 8];
  __shared__ float    sMax[16];
  __shared__ float    sRcp[16];

  const int qt   = blockIdx.x;
  const int bh   = blockIdx.y;
  const int b    = bh >> 3;
  const int h    = bh & 7;
  const int tid  = threadIdx.x;
  const int lane = tid & 31;
  const int wave = tid >> 5;
  const int n    = lane & 15;           // fragment column (B/C) == A row
  const int hi   = lane >> 4;

  const float scale = 0.125f;
  const size_t baseQ = (size_t)b * SEQ * DM + (size_t)h * DH;
  const _Float16* Kh  = ws + WS_KH  + ((size_t)b * SEQP) * DM + (size_t)h * DH;
  const _Float16* Vt  = ws + WS_VT  + (size_t)bh * DH * SPAD;
  const _Float16* RKh = ws + WS_RKH;
  const _Float16* RVt = ws + WS_RVT;

  // ---------- Phase 0: stage Q tile (f16); zero distance accumulator -------
  for (int idx = tid; idx < 16 * DH; idx += 128) {
    int m = idx >> 6, d = idx & 63;
    int qrow = qt * 16 + m;
    float v = (qrow < SEQ) ? Q[baseQ + (size_t)qrow * DM + d] : 0.0f;
    sQ[m * DH + d] = (_Float16)v;
  }
  for (int idx = tid; idx < 16 * NDIST; idx += 128) sD[idx] = 0.0f;
  __syncthreads();

  // A fragments of the Q tile (K=0..31 and K=32..63) — invariant all phases
  v16h qa0, qa1;
  #pragma unroll
  for (int e = 0; e < 16; ++e) {
    int koff = hi * 16 + e;
    qa0[e] = sQ[n * DH + koff];
    qa1[e] = sQ[n * DH + 32 + koff];
  }

  // ---------- Phase 1: P = Qtile @ rel_k^T ----------------------------------
  for (int ct = wave; ct < NDIST / 16; ct += NW) {
    const int dist = ct * 16 + n;
    const _Float16* rp = RKh + (size_t)dist * DH + hi * 16;
    v8f acc = {};
    acc = wmma16(qa0, *(const v16h*)(rp),      acc);
    acc = wmma16(qa1, *(const v16h*)(rp + 32), acc);
    #pragma unroll
    for (int r = 0; r < 8; ++r)
      sP[(r + 8 * hi) * NDIST + ct * 16 + n] = (_Float16)acc[r];
  }
  __syncthreads();

  // ---------- Phase 2: scores = QK^T*scale + gather(P, circ_dist) ----------
  for (int kt = wave; kt < KTI; kt += NW) {
    const int key = kt * 16 + n;
    const _Float16* kp = Kh + (size_t)key * DM + hi * 16;
    if (kt + NW < KTI)   // prefetch next key tile (global_prefetch_b8)
      __builtin_prefetch(Kh + (size_t)(key + NW * 16) * DM + hi * 16, 0, 1);
    v8f acc = {};
    acc = wmma16(qa0, *(const v16h*)(kp),      acc);
    acc = wmma16(qa1, *(const v16h*)(kp + 32), acc);
    #pragma unroll
    for (int r = 0; r < 8; ++r) {
      int m = r + 8 * hi;
      int qrow = qt * 16 + m;
      float sc;
      if (key < SEQ) {
        int d = circ_dist(qrow, key);
        sc = acc[r] * scale + (float)sP[m * NDIST + d];
      } else {
        sc = -1.0e30f;
      }
      sS[m * SPAD + kt * 16 + n] = sc;
    }
  }
  __syncthreads();

  // ---------- Phase 3: row softmax (normalization deferred) -----------------
  {
    const int row = tid >> 3, sub = tid & 7;
    const int j0 = sub * 128;
    float* rowS = sS + row * SPAD;

    float mx = -1.0e30f;
    for (int j = j0; j < j0 + 128 && j < SEQ; j += 4) {
      v4f v = *(const v4f*)(rowS + j);
      mx = fmaxf(mx, fmaxf(fmaxf(v.x, v.y), fmaxf(v.z, v.w)));
    }
    sRed[row * 8 + sub] = mx;
    __syncthreads();
    if (sub == 0) {
      float m2 = sRed[row * 8];
      #pragma unroll
      for (int u = 1; u < 8; ++u) m2 = fmaxf(m2, sRed[row * 8 + u]);
      sMax[row] = m2;
    }
    __syncthreads();
    const float rm = sMax[row];
    float sum = 0.0f;
    for (int j = j0; j < j0 + 128; j += 4) {
      v4f v = *(const v4f*)(rowS + j);
      v4f e;
      if (j < SEQ) {   // SEQ is a multiple of 4: whole float4 valid or dead
        e.x = __expf(v.x - rm); e.y = __expf(v.y - rm);
        e.z = __expf(v.z - rm); e.w = __expf(v.w - rm);
      } else {
        e = (v4f){0.0f, 0.0f, 0.0f, 0.0f};
      }
      *(v4f*)(rowS + j) = e;
      v4h h4; h4.x = (_Float16)e.x; h4.y = (_Float16)e.y;
      h4.z = (_Float16)e.z; h4.w = (_Float16)e.w;
      *(v4h*)(sA + row * SPAD + j) = h4;
      sum += e.x + e.y + e.z + e.w;
    }
    sRed[row * 8 + sub] = sum;
    __syncthreads();
    if (sub == 0) {
      float s2 = 0.0f;
      #pragma unroll
      for (int u = 0; u < 8; ++u) s2 += sRed[row * 8 + u];
      sRcp[row] = 1.0f / s2;
    }
    __syncthreads();
  }

  // ---------- Phase 4: scatter attn mass into distance buckets --------------
  for (int idx = tid; idx < 16 * SPAD; idx += 128) {
    int m = idx >> 10, j = idx & (SPAD - 1);
    if (j < SEQ) {
      int qrow = qt * 16 + m;
      int d = circ_dist(qrow, j);
      atomicAdd(&sD[m * NDIST + d], sS[idx]);   // ds_add_f32
    }
  }
  __syncthreads();

  // ---------- Phase 5: out = (attn@V + A_dist@rel_v) * rcp ------------------
  {
    const int dbase = wave * 16;                 // each wave owns 16 head dims
    const _Float16* vcol  = Vt  + (size_t)(dbase + n) * SPAD  + hi * 16;
    const _Float16* rvcol = RVt + (size_t)(dbase + n) * NDIST + hi * 16;
    v8f acc = {};

    for (int kt2 = 0; kt2 < SPAD / 32; ++kt2) {
      const int kb = kt2 * 32;
      v16h a;
      #pragma unroll
      for (int e = 0; e < 16; ++e) {
        int koff = hi * 16 + e;
        a[e] = sA[n * SPAD + kb + koff];
      }
      acc = wmma16(a, *(const v16h*)(vcol + kb), acc);
    }
    for (int rt = 0; rt < NDIST / 32; ++rt) {
      const int kb = rt * 32;
      v16h a;
      #pragma unroll
      for (int e = 0; e < 16; ++e) {
        int koff = hi * 16 + e;
        a[e] = (_Float16)sD[n * NDIST + kb + koff];
      }
      acc = wmma16(a, *(const v16h*)(rvcol + kb), acc);
    }
    #pragma unroll
    for (int r = 0; r < 8; ++r) {
      int m = r + 8 * hi;
      int qrow = qt * 16 + m;
      if (qrow < SEQ)
        Out[baseQ + (size_t)qrow * DM + dbase + n] = acc[r] * sRcp[m];
    }
  }
}

extern "C" void kernel_launch(void* const* d_in, const int* in_sizes, int n_in,
                              void* d_out, int out_size, void* d_ws, size_t ws_size,
                              hipStream_t stream) {
  (void)in_sizes; (void)n_in; (void)out_size; (void)ws_size;
  const float* Q  = (const float*)d_in[0];
  const float* K  = (const float*)d_in[1];
  const float* V  = (const float*)d_in[2];
  const float* RK = (const float*)d_in[3];
  const float* RV = (const float*)d_in[4];
  float* Out = (float*)d_out;
  _Float16* ws = (_Float16*)d_ws;   // needs 16 MB

  // prep: f16 operand copies / transposes
  prep_k_kernel  <<<(8 * SEQP * DM) / 256, 256, 0, stream>>>(K, ws);
  prep_vt_kernel <<<(64 * DH * SPAD) / 256, 256, 0, stream>>>(V, ws);
  prep_rel_kernel<<<(2 * NDIST * DH) / 256, 256, 0, stream>>>(RK, RV, ws);

  // fused attention
  dim3 grid(QTN, 64);
  circ_rel_attn_kernel<<<grid, dim3(128), 0, stream>>>(Q, ws, Out);
}
